// _ResidualBlock_13383118094729
// MI455X (gfx1250) — matrix operand order repaired and checked by reading, not compiled, not run on hardware
//
#include <hip/hip_runtime.h>
#include <math.h>

typedef __attribute__((ext_vector_type(16))) __bf16 v16bf;
typedef __attribute__((ext_vector_type(8)))  float  v8f;

#define NB       8
#define CH       512
#define TOK_TILE 16
#define N_TILE   128
#define NSUB     (N_TILE / 16)      // 8 C-tiles per wave
#define K_STEP   32
#define K_ITERS  (CH / K_STEP)      // 16

// ---------- compile-time Cl(3,0) Cayley sign tables -------------------------
struct Tab { float c[8][8]; float cr[8][8]; };
constexpr int gp_cnt(int a, int b) {
  int t = a >> 1, c = 0;
  while (t) { int x = t & b; while (x) { c += x & 1; x >>= 1; } t >>= 1; }
  return c & 1;
}
constexpr int grade(int k) { int c = 0; while (k) { c += k & 1; k >>= 1; } return c; }
constexpr Tab mk_tab() {
  Tab t{};
  for (int a = 0; a < 8; ++a)
    for (int b = 0; b < 8; ++b) {
      float s = gp_cnt(a, b) ? -1.f : 1.f;
      int g = grade(b);
      float rs = ((g * (g - 1) / 2) & 1) ? -1.f : 1.f;  // reversion sign of blade b
      t.c[a][b] = s;
      t.cr[a][b] = s * rs;  // sign for right-multiplication by ~R folded in
    }
  return t;
}
constexpr Tab TAB = mk_tab();

// full geometric product d = a * b (used only in tiny prep kernel)
__device__ __forceinline__ void gp_fwd(const float* a, const float* b, float* d) {
#pragma unroll
  for (int k = 0; k < 8; ++k) d[k] = 0.f;
#pragma unroll
  for (int i = 0; i < 8; ++i)
#pragma unroll
    for (int j = 0; j < 8; ++j)
      d[i ^ j] = fmaf(TAB.c[i][j] * a[i], b[j], d[i ^ j]);
}

// v <- L * v * ~L for an EVEN rotor L = (q.x, e12:q.y, e13:q.z, e23:q.w). 64 FMAs.
__device__ __forceinline__ void sandwich_even(const float4 q, float* v) {
  constexpr int EV[4] = {0, 3, 5, 6};
  const float a[4] = {q.x, q.y, q.z, q.w};
  float t[8];
#pragma unroll
  for (int k = 0; k < 8; ++k) t[k] = 0.f;
#pragma unroll
  for (int e = 0; e < 4; ++e) {
    const int i = EV[e];
#pragma unroll
    for (int j = 0; j < 8; ++j) t[i ^ j] = fmaf(TAB.c[i][j] * a[e], v[j], t[i ^ j]);
  }
#pragma unroll
  for (int k = 0; k < 8; ++k) v[k] = 0.f;
#pragma unroll
  for (int i = 0; i < 8; ++i)
#pragma unroll
    for (int e = 0; e < 4; ++e) {
      const int j = EV[e];
      v[i ^ j] = fmaf(TAB.cr[i][j] * t[i], a[e], v[i ^ j]);
    }
}

__device__ __forceinline__ unsigned short f2bf(float f) {
  unsigned u = __float_as_uint(f);
  u += 0x7FFFu + ((u >> 16) & 1u);   // round to nearest even
  return (unsigned short)(u >> 16);
}

// ---------- prep kernels ----------------------------------------------------
__global__ void prep_weights(const float* __restrict__ w, unsigned short* __restrict__ wbf, int n) {
  int i = blockIdx.x * blockDim.x + threadIdx.x;
  if (i < n) wbf[i] = f2bf(w[i]);
}

// Compose the 4 sequential rotor sandwiches into one even rotor per channel:
// L[c] = R3*R2*R1*R0  (sandwiches compose exactly: R1(R0 x ~R0)~R1 = (R1R0) x ~(R1R0))
__global__ void prep_rotors(const float* __restrict__ th, float* __restrict__ rq) {
  int c = blockIdx.x * blockDim.x + threadIdx.x;
  if (c >= CH) return;
  float tot[8], tmp[8], rv[8];
#pragma unroll
  for (int r = 0; r < 4; ++r) {
    const float* tp = th + ((size_t)r * CH + c) * 4;
    float t0 = tp[0], t1 = tp[1], t2 = tp[2], t3 = tp[3];
    float inv = 1.f / sqrtf(t0 * t0 + t1 * t1 + t2 * t2 + t3 * t3 + 1e-6f);
    float o[8] = {t0 * inv, 0.f, 0.f, t1 * inv, 0.f, t2 * inv, t3 * inv, 0.f};
    if (r == 0) {
#pragma unroll
      for (int k = 0; k < 8; ++k) tot[k] = o[k];
    } else {
#pragma unroll
      for (int k = 0; k < 8; ++k) rv[k] = o[k];
      gp_fwd(rv, tot, tmp);          // tot <- R_r * tot
#pragma unroll
      for (int k = 0; k < 8; ++k) tot[k] = tmp[k];
    }
  }
  float4* o4 = (float4*)(rq + (size_t)c * 4);
  *o4 = make_float4(tot[0], tot[3], tot[5], tot[6]);   // even components only
}

// ---------- fused main kernel ----------------------------------------------
// grid: (BT/16, CH/128). block: 256 threads = 8 waves; wave w computes blade w.
__global__ __launch_bounds__(256) void clifford_block(
    const float* __restrict__ x, const float* __restrict__ gamma,
    const float* __restrict__ lin_b, const unsigned short* __restrict__ wbf,
    const float* __restrict__ rq, const float* __restrict__ blade_w,
    float* __restrict__ out) {
  __shared__ __align__(32) unsigned short a_lds[2][NB * 32 * 16];  // 2 x 8 KB (double buffer)
  __shared__ float phi_lds[TOK_TILE * N_TILE];                     // 8 KB
  __shared__ float mv_lds[TOK_TILE * N_TILE * NB];                 // 64 KB

  const int tid = threadIdx.x;
  const int wk = tid >> 5;       // wave id == blade index
  const int lane = tid & 31;
  const int ln15 = lane & 15;
  const int hi = lane >> 4;      // 0 or 1 (half-wave)
  const int tok0 = blockIdx.x * TOK_TILE;
  const int nbase = blockIdx.y * N_TILE;

  v8f acc[NSUB] = {};
  // B layout: lane = column, half h -> K = h + 16*(lane>=16): 32B contiguous per lane.
  // One base pointer; per-subtile offset s*16*CH folds into the 24-bit IOFFSET.
  const unsigned short* bbase =
      wbf + ((size_t)wk * CH + nbase + ln15) * CH + (hi << 4);

  // --- staging helper (layernorm + gamma -> bf16, scattered into A-frag layout)
  auto stage_a = [&](unsigned short* buf, int kk) {
#pragma unroll
    for (int pp = 0; pp < 2; ++pp) {
      int p = tid * 2 + pp;            // 512 multivectors: 16 tok x 32 ch
      int m = p >> 5;
      int c = p & 31;
      const float* xp = x + ((size_t)(tok0 + m) * CH + (kk + c)) * NB;
      float4 a0 = *(const float4*)xp;
      float4 a1 = *(const float4*)(xp + 4);
      if (kk + K_STEP < CH)            // prefetch next K-step's rows (in-bounds by guard)
        __builtin_prefetch((const void*)(xp + K_STEP * NB), 0, 3);
      float ss = a0.x * a0.x + a0.y * a0.y + a0.z * a0.z + a0.w * a0.w +
                 a1.x * a1.x + a1.y * a1.y + a1.z * a1.z + a1.w * a1.w;
      float sc = gamma[kk + c] / sqrtf(ss + 1e-6f);
      // A-matrix 16x32 bf16: lanes 0-15 hold K in [0,16), lanes 16-31 hold K in [16,32)
      int wl = m + ((c >> 4) << 4);
      int h = c & 15;
      int base = wl * 16 + h;
      buf[base + 0 * 512] = f2bf(a0.x * sc);
      buf[base + 1 * 512] = f2bf(a0.y * sc);
      buf[base + 2 * 512] = f2bf(a0.z * sc);
      buf[base + 3 * 512] = f2bf(a0.w * sc);
      buf[base + 4 * 512] = f2bf(a1.x * sc);
      buf[base + 5 * 512] = f2bf(a1.y * sc);
      buf[base + 6 * 512] = f2bf(a1.z * sc);
      buf[base + 7 * 512] = f2bf(a1.w * sc);
    }
  };

  stage_a(a_lds[0], 0);
  __syncthreads();
  for (int it = 0; it < K_ITERS; ++it) {
    const int kk = it * K_STEP;
    v16bf af = *(const v16bf*)(&a_lds[it & 1][(wk * 32 + lane) * 16]);
    if (it + 1 < K_ITERS) stage_a(a_lds[(it + 1) & 1], kk + K_STEP);
#pragma unroll
    for (int s = 0; s < NSUB; ++s) {
      v16bf bf = *(const v16bf*)(bbase + (size_t)s * 16 * CH + kk);
      acc[s] = __builtin_amdgcn_wmma_f32_16x16x32_bf16(
          false, af, false, bf, (short)0, acc[s], false, false);
    }
    __syncthreads();
  }

  // --- bias add (before GELU gate, per reference)
#pragma unroll
  for (int s = 0; s < NSUB; ++s) {
    float bias = lin_b[(size_t)(nbase + s * 16 + ln15) * NB + wk];
#pragma unroll
    for (int r = 0; r < 8; ++r) acc[s][r] += bias;
  }
  // --- wave 0 (scalar blade) publishes Gaussian-CDF gate phi
  if (wk == 0) {
#pragma unroll
    for (int s = 0; s < NSUB; ++s)
#pragma unroll
      for (int r = 0; r < 8; ++r) {
        int m = r + hi * 8;
        phi_lds[m * N_TILE + s * 16 + ln15] =
            0.5f * (1.f + erff(acc[s][r] * 0.70710678118654752f));
      }
  }
  __syncthreads();
  // --- gate all blades, assemble multivector tile in LDS
#pragma unroll
  for (int s = 0; s < NSUB; ++s)
#pragma unroll
    for (int r = 0; r < 8; ++r) {
      int m = r + hi * 8;
      int o = s * 16 + ln15;
      mv_lds[(m * N_TILE + o) * NB + wk] = acc[s][r] * phi_lds[m * N_TILE + o];
    }
  __syncthreads();

  // --- composed rotor sandwich + blade gate + residual (per-thread multivectors)
  int ol = tid & (N_TILE - 1);   // channel within tile (0..127)
  int g = tid >> 7;              // token group (8 tokens each)
  int ch = nbase + ol;
  float4 q = *(const float4*)(rq + (size_t)ch * 4);
  float gate[8];
#pragma unroll
  for (int k = 0; k < 8; ++k)
    gate[k] = 1.f / (1.f + expf(-blade_w[(size_t)ch * NB + k]));

#pragma unroll
  for (int u = 0; u < 8; ++u) {
    int m = g * 8 + u;
    float v[8];
#pragma unroll
    for (int k = 0; k < 8; ++k) v[k] = mv_lds[(m * N_TILE + ol) * NB + k];
    sandwich_even(q, v);         // all 4 rotor sandwiches, composed
    size_t base = ((size_t)(tok0 + m) * CH + ch) * NB;
    float4 x0 = *(const float4*)(x + base);
    float4 x1 = *(const float4*)(x + base + 4);
    float o8[8] = {x0.x, x0.y, x0.z, x0.w, x1.x, x1.y, x1.z, x1.w};
#pragma unroll
    for (int k = 0; k < 8; ++k) o8[k] += v[k] * gate[k];
    *(float4*)(out + base)     = make_float4(o8[0], o8[1], o8[2], o8[3]);
    *(float4*)(out + base + 4) = make_float4(o8[4], o8[5], o8[6], o8[7]);
  }
}

// ---------- host entry ------------------------------------------------------
extern "C" void kernel_launch(void* const* d_in, const int* in_sizes, int n_in,
                              void* d_out, int out_size, void* d_ws, size_t ws_size,
                              hipStream_t stream) {
  const float* x       = (const float*)d_in[0];  // [2,4096,512,8]
  const float* gamma   = (const float*)d_in[1];  // [512]
  const float* lin_w   = (const float*)d_in[2];  // [8,512,512]
  const float* lin_b   = (const float*)d_in[3];  // [512,8]
  const float* rtheta  = (const float*)d_in[4];  // [4,512,4]
  const float* blade_w = (const float*)d_in[5];  // [512,8]
  float* out = (float*)d_out;

  // workspace: bf16 weights (4 MiB) + composed even rotors (8 KiB)
  unsigned short* wbf = (unsigned short*)d_ws;
  float* rq = (float*)((char*)d_ws + (size_t)NB * CH * CH * sizeof(unsigned short));

  const int nw = NB * CH * CH;  // 2,097,152
  prep_weights<<<(nw + 255) / 256, 256, 0, stream>>>(lin_w, wbf, nw);
  prep_rotors<<<(CH + 255) / 256, 256, 0, stream>>>(rtheta, rq);

  dim3 grid(8192 / TOK_TILE, CH / N_TILE);  // (512, 4)
  clifford_block<<<grid, 256, 0, stream>>>(x, gamma, lin_b, wbf, rq, blade_w, out);
}
